// RareVariantGNN_38062000177239
// MI455X (gfx1250) — matrix operand order repaired and checked by reading, not compiled, not run on hardware
//
#include <hip/hip_runtime.h>
#include <hip/hip_bf16.h>

typedef __attribute__((ext_vector_type(16))) __bf16 v16bf;
typedef __attribute__((ext_vector_type(8)))  __bf16 v8bf;
typedef __attribute__((ext_vector_type(4)))  __bf16 v4bf;
typedef __attribute__((ext_vector_type(8)))  float  v8f;
typedef __attribute__((ext_vector_type(4)))  float  fv4;

__device__ __forceinline__ float gelu_f(float x) {
  return 0.5f * x * (1.f + erff(x * 0.70710678118654752f));
}

__device__ __forceinline__ void atomicMaxF(float* addr, float v) {
  if (v >= 0.f) atomicMax((int*)addr, __float_as_int(v));
  else          atomicMin((unsigned int*)addr, __float_as_uint(v));
}

// ------------------------------------------------------------------
// WMMA bf16 GEMM:  C[M,N] = act( A[M,K] * W + bias )
// W pre-packed bf16, layout Bt[N][K]. Requires N%64==0, K%32==0.
// Block tile 128x64x32, 8 wave32 waves (4x2), wave tile 32x32 = 2x2 WMMA.
// A staged fp32->bf16 via batched b128 loads (uniform fast path);
// B staged via global_load_async_to_lds_b128 (ASYNCcnt DMA path).
// ------------------------------------------------------------------
#define BM 128
#define BN 64
#define BK 32
#define LDA 40   // padded LDS row stride (bf16) -> conflict-free b128 frags
#define LDB 40

__global__ __launch_bounds__(256)
void gemm_bf16_wmma(const float* __restrict__ A, const __bf16* __restrict__ Bt,
                    const float* __restrict__ bias,
                    float* __restrict__ Cf, __bf16* __restrict__ Cb,
                    int M, int N, int K, int act)
{
  __shared__ alignas(16) __bf16 sA[BM * LDA];
  __shared__ alignas(16) __bf16 sB[BN * LDB];

  const int tid  = threadIdx.x;
  const int lane = tid & 31;
  const int wave = tid >> 5;
  const int m0 = blockIdx.x * BM;
  const int n0 = blockIdx.y * BN;
  const int mw = (wave >> 1) * 32;      // wave M offset in block tile
  const int nw = (wave & 1)  * 32;      // wave N offset
  const int half = lane >> 4;           // 0: lanes 0-15, 1: lanes 16-31
  const int l16  = lane & 15;

  const bool fullM = (m0 + BM <= M);    // block-uniform
  // A staging map: 1024 float4 per tile, 4 per thread
  const int ar  = tid >> 3;             // base row 0..31 (+32*q)
  const int ac4 = tid & 7;              // float4 index within 32-col row
  // B staging map: 256 x 16B chunks, 1 per thread
  const int br = tid >> 2;              // 0..63
  const int bc = (tid & 3) * 8;         // bf16 col base
  const unsigned sB_dst =
      (unsigned)(size_t)(&sB[0]) + (unsigned)((br * LDB + bc) * 2);
  const __bf16* bSrcBase = Bt + (size_t)(n0 + br) * K + bc;
  const float* aBase = A + (size_t)(m0 + ar) * K + ac4 * 4;

  v8f acc[2][2] = {};

  for (int k0 = 0; k0 < K; k0 += BK) {
    // ---- B tile: async DMA straight into LDS (already bf16) ----
    asm volatile("global_load_async_to_lds_b128 %0, %1, off"
                 :: "v"(sB_dst), "v"(bSrcBase + k0) : "memory");

    // ---- A tile: batched b128 loads -> packed bf16 b64 stores ----
    fv4 av[4];
    if (fullM) {                        // uniform branch: no predication
#pragma unroll
      for (int q = 0; q < 4; ++q)
        av[q] = *(const fv4*)(aBase + (size_t)(32 * q) * K + k0);
    } else {
#pragma unroll
      for (int q = 0; q < 4; ++q) {
        int gm = m0 + ar + 32 * q;
        av[q] = (gm < M) ? *(const fv4*)(aBase + (size_t)(32 * q) * K + k0)
                         : (fv4){0.f, 0.f, 0.f, 0.f};
      }
    }
#pragma unroll
    for (int q = 0; q < 4; ++q) {
      v4bf b;
#pragma unroll
      for (int i = 0; i < 4; ++i) b[i] = (__bf16)av[q][i];
      *(v4bf*)&sA[(ar + 32 * q) * LDA + ac4 * 4] = b;
    }

    asm volatile("s_wait_asynccnt 0x0" ::: "memory");
    __syncthreads();

    // ---- fragments per documented 16-bit A/B VGPR layouts ----
    union Frag { v16bf v; v8bf h[2]; } afr[2], bfr[2];
#pragma unroll
    for (int mi = 0; mi < 2; ++mi) {
      const __bf16* p = &sA[(mw + mi * 16 + l16) * LDA + half * 8];
      afr[mi].h[0] = *(const v8bf*)p;
      afr[mi].h[1] = *(const v8bf*)(p + 16);
    }
#pragma unroll
    for (int ni = 0; ni < 2; ++ni) {
      const __bf16* p = &sB[(nw + ni * 16 + l16) * LDB + half * 8];
      bfr[ni].h[0] = *(const v8bf*)p;
      bfr[ni].h[1] = *(const v8bf*)(p + 16);
    }
#pragma unroll
    for (int mi = 0; mi < 2; ++mi)
#pragma unroll
      for (int ni = 0; ni < 2; ++ni)
        acc[mi][ni] = __builtin_amdgcn_wmma_f32_16x16x32_bf16(
            false, afr[mi].v, false, bfr[ni].v, (short)0, acc[mi][ni],
            false, false);
    __syncthreads();
  }

  // ---- epilogue: C layout = VGPR i -> row half*8+i, col = lane&15 ----
#pragma unroll
  for (int mi = 0; mi < 2; ++mi) {
#pragma unroll
    for (int ni = 0; ni < 2; ++ni) {
      int col = n0 + nw + ni * 16 + l16;
      float bv = bias ? bias[col] : 0.f;
#pragma unroll
      for (int i = 0; i < 8; ++i) {
        int row = m0 + mw + mi * 16 + half * 8 + i;
        if (row >= M) continue;
        float v = acc[mi][ni][i] + bv;
        if (act == 1) v = gelu_f(v);
        if (Cb) Cb[(size_t)row * N + col] = (__bf16)v;
        else    Cf[(size_t)row * N + col] = v;
      }
    }
  }
}

// ------------------- weight pack: fp32 [K][N] -> bf16 [N][K] -------------
__global__ void pack_wt(const float* __restrict__ W, __bf16* __restrict__ out,
                        int K, int N) {
  size_t idx = (size_t)blockIdx.x * blockDim.x + threadIdx.x;
  size_t tot = (size_t)K * N;
  if (idx >= tot) return;
  int n = (int)(idx / K), k = (int)(idx % K);
  out[idx] = (__bf16)W[(size_t)k * N + n];
}

// ------------------- fills -------------------
__global__ void fill_f32(float* p, float v, size_t n) {
  size_t i = (size_t)blockIdx.x * blockDim.x + threadIdx.x;
  size_t st = (size_t)gridDim.x * blockDim.x;
  for (; i < n; i += st) p[i] = v;
}

__global__ void bcast_fill(float* p, const float* __restrict__ row,
                           unsigned mask, size_t n) {
  size_t i = (size_t)blockIdx.x * blockDim.x + threadIdx.x;
  size_t st = (size_t)gridDim.x * blockDim.x;
  for (; i < n; i += st) p[i] = row[i & mask];
}

// ------------------- per-row LayerNorm + GELU (one wave per row) ---------
__global__ __launch_bounds__(256)
void ln_gelu(float* x, int rows, int C, const float* __restrict__ w,
             const float* __restrict__ b) {
  int row = blockIdx.x * 8 + (threadIdx.x >> 5);
  int lane = threadIdx.x & 31;
  if (row >= rows) return;
  float* p = x + (size_t)row * C;
  float s = 0.f, s2 = 0.f;
  for (int i = lane; i < C; i += 32) { float v = p[i]; s += v; s2 += v * v; }
#pragma unroll
  for (int m = 16; m; m >>= 1) {
    s  += __shfl_xor(s, m, 32);
    s2 += __shfl_xor(s2, m, 32);
  }
  float mean = s / (float)C;
  float var  = s2 / (float)C - mean * mean;
  float rstd = rsqrtf(var + 1e-5f);
  for (int i = lane; i < C; i += 32) {
    float v = (p[i] - mean) * rstd * w[i] + b[i];
    p[i] = gelu_f(v);
  }
}

// ------------------- GATv2 edge kernels ------------------------------
// xl/xr bf16 [N][512] head-major. One wave per edge; lane owns 16
// contiguous channels (two b128 loads); head = lane>>3 is lane-uniform.
__global__ __launch_bounds__(256)
void edge_score(const __bf16* __restrict__ xl, const __bf16* __restrict__ xr,
                const float* __restrict__ att, const int* __restrict__ src,
                const int* __restrict__ dst, int E, int ET,
                float* __restrict__ score, float* __restrict__ smax) {
  int e = (int)(((size_t)blockIdx.x * blockDim.x + threadIdx.x) >> 5);
  int lane = threadIdx.x & 31;
  if (e >= ET) return;
  int s = (e < E) ? src[e] : (e - E);
  int d = (e < E) ? dst[e] : (e - E);
  const __bf16* pl = xl + (size_t)s * 512 + lane * 16;
  const __bf16* pr = xr + (size_t)d * 512 + lane * 16;
  v8bf l0 = *(const v8bf*)pl, l1 = *(const v8bf*)(pl + 8);
  v8bf r0 = *(const v8bf*)pr, r1 = *(const v8bf*)(pr + 8);
  fv4 a0 = *(const fv4*)(att + lane * 16);
  fv4 a1 = *(const fv4*)(att + lane * 16 + 4);
  fv4 a2 = *(const fv4*)(att + lane * 16 + 8);
  fv4 a3 = *(const fv4*)(att + lane * 16 + 12);
  float part = 0.f;
#pragma unroll
  for (int j = 0; j < 4; ++j) {
    float v = (float)l0[j] + (float)r0[j];
    part += ((v > 0.f) ? v : 0.2f * v) * a0[j];
  }
#pragma unroll
  for (int j = 0; j < 4; ++j) {
    float v = (float)l0[j + 4] + (float)r0[j + 4];
    part += ((v > 0.f) ? v : 0.2f * v) * a1[j];
  }
#pragma unroll
  for (int j = 0; j < 4; ++j) {
    float v = (float)l1[j] + (float)r1[j];
    part += ((v > 0.f) ? v : 0.2f * v) * a2[j];
  }
#pragma unroll
  for (int j = 0; j < 4; ++j) {
    float v = (float)l1[j + 4] + (float)r1[j + 4];
    part += ((v > 0.f) ? v : 0.2f * v) * a3[j];
  }
  // reduce within each 8-lane head group
  part += __shfl_xor(part, 1, 32);
  part += __shfl_xor(part, 2, 32);
  part += __shfl_xor(part, 4, 32);
  if ((lane & 7) == 0) {
    int h = lane >> 3;
    score[(size_t)e * 4 + h] = part;
    atomicMaxF(&smax[(size_t)d * 4 + h], part);
  }
}

__global__ void edge_exp(float* __restrict__ score,
                         const float* __restrict__ smax,
                         float* __restrict__ denom,
                         const int* __restrict__ dst, int E, int ET) {
  size_t idx = (size_t)blockIdx.x * blockDim.x + threadIdx.x;
  if (idx >= (size_t)ET * 4) return;
  int e = (int)(idx >> 2), h = (int)(idx & 3);
  int d = (e < E) ? dst[e] : (e - E);
  float a = expf(score[idx] - smax[(size_t)d * 4 + h]);
  score[idx] = a;
  atomicAdd(&denom[(size_t)d * 4 + h], a);
}

__global__ __launch_bounds__(256)
void edge_scatter(const __bf16* __restrict__ xl, const float* __restrict__ score,
                  const float* __restrict__ denom, const int* __restrict__ src,
                  const int* __restrict__ dst, int E, int ET,
                  float* __restrict__ agg) {
  int e = (int)(((size_t)blockIdx.x * blockDim.x + threadIdx.x) >> 5);
  int lane = threadIdx.x & 31;
  if (e >= ET) return;
  int s = (e < E) ? src[e] : (e - E);
  int d = (e < E) ? dst[e] : (e - E);
  int h = lane >> 3;   // lane-uniform head
  float alpha = score[(size_t)e * 4 + h] /
                (denom[(size_t)d * 4 + h] + 1e-16f);
  const __bf16* pl = xl + (size_t)s * 512 + lane * 16;
  v8bf l0 = *(const v8bf*)pl, l1 = *(const v8bf*)(pl + 8);
  float* po = agg + (size_t)d * 512 + lane * 16;
#pragma unroll
  for (int j = 0; j < 8; ++j) atomicAdd(&po[j], alpha * (float)l0[j]);
#pragma unroll
  for (int j = 0; j < 8; ++j) atomicAdd(&po[j + 8], alpha * (float)l1[j]);
}

// ------------------- graph-norm (global mean/var) ---------------------
__global__ void gstats(const float* __restrict__ x, size_t n, float* stats) {
  size_t i = (size_t)blockIdx.x * blockDim.x + threadIdx.x;
  size_t st = (size_t)gridDim.x * blockDim.x;
  float s = 0.f, s2 = 0.f;
  for (; i < n; i += st) { float v = x[i]; s += v; s2 += v * v; }
#pragma unroll
  for (int m = 16; m; m >>= 1) {
    s  += __shfl_xor(s, m, 32);
    s2 += __shfl_xor(s2, m, 32);
  }
  if ((threadIdx.x & 31) == 0) {
    atomicAdd(&stats[0], s);
    atomicAdd(&stats[1], s2);
  }
}

__global__ void gfinalize(float* stats, float invN) {
  if (threadIdx.x == 0 && blockIdx.x == 0) {
    float m = stats[0] * invN;
    float v = stats[1] * invN - m * m;
    stats[2] = m;
    stats[3] = rsqrtf(v + 1e-5f);
  }
}

__global__ void gnorm_apply(const float* __restrict__ pj,
                            const float* __restrict__ stats,
                            const float* __restrict__ w,
                            const float* __restrict__ b,
                            float* __restrict__ xcur, float* __restrict__ xcat,
                            int n, int layer) {
  size_t idx = (size_t)blockIdx.x * blockDim.x + threadIdx.x;
  size_t tot = (size_t)n * 128;
  size_t st = (size_t)gridDim.x * blockDim.x;
  float mean = stats[2], rstd = stats[3];
  for (; idx < tot; idx += st) {
    int node = (int)(idx >> 7), c = (int)(idx & 127);
    float v = (pj[idx] - mean) * rstd * w[c] + b[c];
    float r = gelu_f(v) + xcur[idx];                 // residual
    xcur[idx] = r;
    xcat[(size_t)node * 384 + (size_t)layer * 128 + c] = r;
  }
}

// ------------------- classifier tail: z2[64] . w3 + b3 -> sigmoid -------
__global__ __launch_bounds__(256)
void clf_final(const float* __restrict__ z2, const float* __restrict__ w3,
               const float* __restrict__ b3, float* __restrict__ out, int n) {
  int row = blockIdx.x * 8 + (threadIdx.x >> 5);
  int lane = threadIdx.x & 31;
  if (row >= n) return;
  const float* p = z2 + (size_t)row * 64;
  float s = p[lane] * w3[lane] + p[lane + 32] * w3[lane + 32];
#pragma unroll
  for (int m = 16; m; m >>= 1) s += __shfl_xor(s, m, 32);
  if (lane == 0) {
    float lg = s + b3[0];
    out[row] = lg;
    out[(size_t)n + row] = 1.f / (1.f + expf(-lg));
  }
}

// =====================================================================
extern "C" void kernel_launch(void* const* d_in, const int* in_sizes, int n_in,
                              void* d_out, int out_size, void* d_ws, size_t ws_size,
                              hipStream_t stream) {
  (void)n_in; (void)out_size; (void)ws_size;
  const int NN = in_sizes[0] / 256;   // 50000 nodes
  const int E  = in_sizes[1] / 2;    // 400000 edges
  const int ET = E + NN;             // + self loops

  const float* x_in = (const float*)d_in[0];
  const int* ei = (const int*)d_in[1];
  const int* srcI = ei;
  const int* dstI = ei + E;

  int pi = 2;
  const float* enc_w1  = (const float*)d_in[pi++];
  const float* enc_b1  = (const float*)d_in[pi++];
  const float* enc_l1w = (const float*)d_in[pi++];
  const float* enc_l1b = (const float*)d_in[pi++];
  const float* enc_w2  = (const float*)d_in[pi++];
  const float* enc_b2  = (const float*)d_in[pi++];
  const float* enc_l2w = (const float*)d_in[pi++];
  const float* enc_l2b = (const float*)d_in[pi++];
  struct GatP { const float *wl,*bl,*wr,*br,*att,*bias,*proj,*nw,*nb; } g[3];
  for (int l = 0; l < 3; ++l) {
    g[l].wl   = (const float*)d_in[pi++];
    g[l].bl   = (const float*)d_in[pi++];
    g[l].wr   = (const float*)d_in[pi++];
    g[l].br   = (const float*)d_in[pi++];
    g[l].att  = (const float*)d_in[pi++];
    g[l].bias = (const float*)d_in[pi++];
    g[l].proj = (const float*)d_in[pi++];
    g[l].nw   = (const float*)d_in[pi++];
    g[l].nb   = (const float*)d_in[pi++];
  }
  const float* c_w1  = (const float*)d_in[pi++];
  const float* c_b1  = (const float*)d_in[pi++];
  const float* c_lnw = (const float*)d_in[pi++];
  const float* c_lnb = (const float*)d_in[pi++];
  const float* c_w2  = (const float*)d_in[pi++];
  const float* c_b2  = (const float*)d_in[pi++];
  const float* c_w3  = (const float*)d_in[pi++];
  const float* c_b3  = (const float*)d_in[pi++];

  // ---- workspace layout ----
  char* wp = (char*)d_ws;
  auto walloc = [&](size_t b) -> void* {
    void* p = (void*)wp; wp += (b + 255) & ~(size_t)255; return p;
  };
  float*  xcur  = (float*) walloc((size_t)NN * 128 * 4);
  float*  xcat  = (float*) walloc((size_t)NN * 384 * 4);
  __bf16* xl    = (__bf16*)walloc((size_t)NN * 512 * 2);
  __bf16* xr    = (__bf16*)walloc((size_t)NN * 512 * 2);
  float*  agg   = (float*) walloc((size_t)NN * 512 * 4); // reused: enc H1, clf z1/z2
  float*  projo = (float*) walloc((size_t)NN * 128 * 4);
  float*  score = (float*) walloc((size_t)ET * 4 * 4);
  float*  smax  = (float*) walloc((size_t)NN * 4 * 4);
  float*  denom = (float*) walloc((size_t)NN * 4 * 4);
  float*  stats = (float*) walloc(256);
  size_t pw_elems = (size_t)256*256 + 256*128 + 3*((size_t)128*512*2 + 512*128)
                  + 384*128 + 128*64;
  __bf16* pwb = (__bf16*)walloc(pw_elems * 2);

  size_t po = 0;
  __bf16* p_e1 = pwb + po; po += (size_t)256 * 256;
  __bf16* p_e2 = pwb + po; po += (size_t)256 * 128;
  __bf16 *p_wl[3], *p_wr[3], *p_pj[3];
  for (int l = 0; l < 3; ++l) {
    p_wl[l] = pwb + po; po += (size_t)128 * 512;
    p_wr[l] = pwb + po; po += (size_t)128 * 512;
    p_pj[l] = pwb + po; po += (size_t)512 * 128;
  }
  __bf16* p_c1 = pwb + po; po += (size_t)384 * 128;
  __bf16* p_c2 = pwb + po; po += (size_t)128 * 64;

  auto pack = [&](const float* W, __bf16* out, int K, int N) {
    size_t tot = (size_t)K * N;
    pack_wt<<<(unsigned)((tot + 255) / 256), 256, 0, stream>>>(W, out, K, N);
  };
  auto gemm = [&](const float* A, const __bf16* Bt, const float* bias,
                  float* Cf, __bf16* Cb, int M, int N, int K, int act) {
    dim3 grid((M + BM - 1) / BM, (N + BN - 1) / BN);
    gemm_bf16_wmma<<<grid, 256, 0, stream>>>(A, Bt, bias, Cf, Cb, M, N, K, act);
  };
  auto fill = [&](float* p, float v, size_t n) {
    size_t b = (n + 255) / 256; if (b > 16384) b = 16384;
    fill_f32<<<(unsigned)b, 256, 0, stream>>>(p, v, n);
  };

  // ---- pack all weights to bf16 [N][K] ----
  pack(enc_w1, p_e1, 256, 256);
  pack(enc_w2, p_e2, 256, 128);
  for (int l = 0; l < 3; ++l) {
    pack(g[l].wl,   p_wl[l], 128, 512);
    pack(g[l].wr,   p_wr[l], 128, 512);
    pack(g[l].proj, p_pj[l], 512, 128);
  }
  pack(c_w1, p_c1, 384, 128);
  pack(c_w2, p_c2, 128, 64);

  // ---- encoder ----
  float* h1 = agg;  // [NN,256] scratch
  gemm(x_in, p_e1, enc_b1, h1, nullptr, NN, 256, 256, 0);
  ln_gelu<<<(NN + 7) / 8, 256, 0, stream>>>(h1, NN, 256, enc_l1w, enc_l1b);
  gemm(h1, p_e2, enc_b2, xcur, nullptr, NN, 128, 256, 0);
  ln_gelu<<<(NN + 7) / 8, 256, 0, stream>>>(xcur, NN, 128, enc_l2w, enc_l2b);

  // ---- GAT layers ----
  const unsigned egBlocks = (unsigned)(((size_t)ET * 32 + 255) / 256);
  for (int l = 0; l < 3; ++l) {
    gemm(xcur, p_wl[l], g[l].bl, nullptr, xl, NN, 512, 128, 0); // bf16 out
    gemm(xcur, p_wr[l], g[l].br, nullptr, xr, NN, 512, 128, 0); // bf16 out

    {   // agg init = broadcast bias (folds "+ bias" into segment-sum)
      size_t n = (size_t)NN * 512;
      size_t b = (n + 255) / 256; if (b > 16384) b = 16384;
      bcast_fill<<<(unsigned)b, 256, 0, stream>>>(agg, g[l].bias, 511u, n);
    }
    fill(smax, -__builtin_huge_valf(), (size_t)NN * 4);
    fill(denom, 0.f, (size_t)NN * 4);
    fill(stats, 0.f, 4);

    edge_score<<<egBlocks, 256, 0, stream>>>(xl, xr, g[l].att, srcI, dstI,
                                             E, ET, score, smax);
    edge_exp<<<(unsigned)(((size_t)ET * 4 + 255) / 256), 256, 0, stream>>>(
        score, smax, denom, dstI, E, ET);
    edge_scatter<<<egBlocks, 256, 0, stream>>>(xl, score, denom, srcI, dstI,
                                               E, ET, agg);

    gemm(agg, p_pj[l], nullptr, projo, nullptr, NN, 128, 512, 0);
    gstats<<<2048, 256, 0, stream>>>(projo, (size_t)NN * 128, stats);
    gfinalize<<<1, 64, 0, stream>>>(stats, 1.f / ((float)NN * 128.f));
    gnorm_apply<<<8192, 256, 0, stream>>>(projo, stats, g[l].nw, g[l].nb,
                                          xcur, xcat, NN, l);
  }

  // ---- classifier ----
  float* z1 = agg;                       // [NN,128]
  float* z2 = agg + (size_t)NN * 128;    // [NN,64]
  gemm(xcat, p_c1, c_b1, z1, nullptr, NN, 128, 384, 0);
  ln_gelu<<<(NN + 7) / 8, 256, 0, stream>>>(z1, NN, 128, c_lnw, c_lnb);
  gemm(z1, p_c2, c_b2, z2, nullptr, NN, 64, 128, 1);   // fused GELU
  clf_final<<<(NN + 7) / 8, 256, 0, stream>>>(z2, c_w3, c_b3, (float*)d_out, NN);
}
